// CenterLoss2_62070867362609
// MI455X (gfx1250) — matrix operand order repaired and checked by reading, not compiled, not run on hardware
//
#include <hip/hip_runtime.h>

// ---- CDNA5 WMMA vector types ----
typedef __attribute__((ext_vector_type(8)))  float   v8f;
typedef __attribute__((ext_vector_type(16))) __bf16  v16bf;
typedef __attribute__((ext_vector_type(8)))  __bf16  v8bf;
typedef __attribute__((ext_vector_type(4)))  __bf16  v4bf;

#define BDIM   4096
#define CDIM   4096
#define DDIM   1024
#define KC     32
#define LDSS   40              // padded LDS row stride in bf16 elems (32 + 8 pad)
#define SCALE  0x1p-25f        // 1 / (2*B*C) = 2^-25 exactly

// -------------------------------------------------------------------------
// Prolog: f2[i] = ||feat_i||^2  -> ws[0..4096)
//         c2[j] = ||centers_j||^2 -> ws[4096..8192)
//         and zero the scalar output.
// One wave32 per row; 8 waves per 256-thread block; 8192 rows total.
// -------------------------------------------------------------------------
__global__ __launch_bounds__(256)
void center_loss_prep(const float* __restrict__ feat,
                      const float* __restrict__ centers,
                      float* __restrict__ ws,
                      float* __restrict__ out) {
    if (blockIdx.x == 0 && threadIdx.x == 0) out[0] = 0.0f;

    const int gwave = (blockIdx.x * 256 + (int)threadIdx.x) >> 5;
    const int lane  = threadIdx.x & 31;

    const float* src = (gwave < BDIM)
                     ? (feat    + (size_t)gwave          * DDIM)
                     : (centers + (size_t)(gwave - BDIM) * DDIM);

    const float4* s4 = (const float4*)src;     // 256 float4 per row
    float s = 0.0f;
    #pragma unroll
    for (int q = 0; q < 8; ++q) {
        float4 v = s4[lane + 32 * q];
        s = fmaf(v.x, v.x, fmaf(v.y, v.y, fmaf(v.z, v.z, fmaf(v.w, v.w, s))));
    }
    #pragma unroll
    for (int off = 16; off; off >>= 1) s += __shfl_xor(s, off);
    if (lane == 0) ws[gwave] = s;
}

// -------------------------------------------------------------------------
// Main: G = feat @ centers^T via bf16 split WMMA (hi*hi + hi*lo + lo*hi),
// fused epilogue  partial += label[i,j]*(f2[i] + c2[j] - 2*G[i,j]).
// 128x128 macro-tile per 256-thread workgroup (8 waves, 16x128 strip/wave).
// -------------------------------------------------------------------------
__global__ __launch_bounds__(256)
void center_loss_main(const float* __restrict__ feat,
                      const float* __restrict__ label,
                      const float* __restrict__ centers,
                      const float* __restrict__ f2,
                      const float* __restrict__ c2,
                      float* __restrict__ out) {
    __shared__ __align__(16) __bf16 sAh[128 * LDSS];
    __shared__ __align__(16) __bf16 sAl[128 * LDSS];
    __shared__ __align__(16) __bf16 sBh[128 * LDSS];
    __shared__ __align__(16) __bf16 sBl[128 * LDSS];

    const int tid  = threadIdx.x;
    const int lane = tid & 31;
    const int wave = tid >> 5;        // 0..7 : 16-row strip of the macro tile
    const int hsel = lane >> 4;       // half-wave select
    const int m    = lane & 15;

    const int rowBase = blockIdx.y * 128;   // feat rows
    const int colBase = blockIdx.x * 128;   // centers rows (= G columns)

    v8f acc[8];
    #pragma unroll
    for (int n = 0; n < 8; ++n)
        acc[n] = (v8f){0.f, 0.f, 0.f, 0.f, 0.f, 0.f, 0.f, 0.f};

    // staging map: 256 threads -> 32 rows x 8 float4-cols per pass, 4 passes
    const int ldRow = tid >> 3;        // 0..31
    const int ldC4  = (tid & 7) * 4;   // 0,4,...,28

    for (int k0 = 0; k0 < DDIM; k0 += KC) {
        __syncthreads();   // protect LDS from previous iteration's readers

        #pragma unroll
        for (int pp = 0; pp < 4; ++pp) {
            const int r = ldRow + pp * 32;
            const float4 fa = *(const float4*)(feat    + (size_t)(rowBase + r) * DDIM + k0 + ldC4);
            const float4 fb = *(const float4*)(centers + (size_t)(colBase + r) * DDIM + k0 + ldC4);
            const float xa[4] = {fa.x, fa.y, fa.z, fa.w};
            const float xb[4] = {fb.x, fb.y, fb.z, fb.w};
            v4bf ah4, al4, bh4, bl4;
            #pragma unroll
            for (int e = 0; e < 4; ++e) {
                __bf16 h = (__bf16)xa[e];
                ah4[e] = h;  al4[e] = (__bf16)(xa[e] - (float)h);
                h = (__bf16)xb[e];
                bh4[e] = h;  bl4[e] = (__bf16)(xb[e] - (float)h);
            }
            *(v4bf*)&sAh[r * LDSS + ldC4] = ah4;
            *(v4bf*)&sAl[r * LDSS + ldC4] = al4;
            *(v4bf*)&sBh[r * LDSS + ldC4] = bh4;
            *(v4bf*)&sBl[r * LDSS + ldC4] = bl4;
        }
        __syncthreads();

        if (k0 + KC < DDIM) {   // warm L1/L2 for the next K slice (global_prefetch)
            __builtin_prefetch(feat    + (size_t)(rowBase + ldRow) * DDIM + k0 + KC + ldC4, 0, 1);
            __builtin_prefetch(centers + (size_t)(colBase + ldRow) * DDIM + k0 + KC + ldC4, 0, 1);
        }

        // A fragment (16x32 bf16): lane<16 holds K {0..7,16..23}, lane>=16 holds {8..15,24..31}
        const int aRow = wave * 16 + m;
        const int ka   = hsel ? 8 : 0;
        v8bf a0h = *(const v8bf*)&sAh[aRow * LDSS + ka];
        v8bf a1h = *(const v8bf*)&sAh[aRow * LDSS + ka + 16];
        v8bf a0l = *(const v8bf*)&sAl[aRow * LDSS + ka];
        v8bf a1l = *(const v8bf*)&sAl[aRow * LDSS + ka + 16];
        v16bf ahi = __builtin_shufflevector(a0h, a1h, 0,1,2,3,4,5,6,7,8,9,10,11,12,13,14,15);
        v16bf alo = __builtin_shufflevector(a0l, a1l, 0,1,2,3,4,5,6,7,8,9,10,11,12,13,14,15);

        #pragma unroll
        for (int n = 0; n < 8; ++n) {
            // B fragment (32x16 bf16): lane = N; lane<16 holds K {0..7,16..23}, lane>=16 holds {8..15,24..31}.
            // B = centers^T  =>  column N of B is row N of centers: contiguous in LDS.
            const int bRow = n * 16 + m;
            const int kb   = hsel ? 8 : 0;
            v8bf b0h = *(const v8bf*)&sBh[bRow * LDSS + kb];
            v8bf b1h = *(const v8bf*)&sBh[bRow * LDSS + kb + 16];
            v8bf b0l = *(const v8bf*)&sBl[bRow * LDSS + kb];
            v8bf b1l = *(const v8bf*)&sBl[bRow * LDSS + kb + 16];
            v16bf bhi = __builtin_shufflevector(b0h, b1h, 0,1,2,3,4,5,6,7,8,9,10,11,12,13,14,15);
            v16bf blo = __builtin_shufflevector(b0l, b1l, 0,1,2,3,4,5,6,7,8,9,10,11,12,13,14,15);

            acc[n] = __builtin_amdgcn_wmma_f32_16x16x32_bf16(false, ahi, false, bhi,
                                                             (short)0, acc[n], false, false);
            acc[n] = __builtin_amdgcn_wmma_f32_16x16x32_bf16(false, ahi, false, blo,
                                                             (short)0, acc[n], false, false);
            acc[n] = __builtin_amdgcn_wmma_f32_16x16x32_bf16(false, alo, false, bhi,
                                                             (short)0, acc[n], false, false);
        }
    }

    // Epilogue. C/D layout: VGPR r, lanes 0-15 -> M=r, lanes 16-31 -> M=r+8; N = lane&15.
    float f2v[8];
    #pragma unroll
    for (int r = 0; r < 8; ++r)
        f2v[r] = f2[rowBase + wave * 16 + r + hsel * 8];

    float part = 0.0f;
    #pragma unroll
    for (int n = 0; n < 8; ++n) {
        const int j   = colBase + n * 16 + m;
        const float c2j = c2[j];
        #pragma unroll
        for (int r = 0; r < 8; ++r) {
            const int i = rowBase + wave * 16 + r + hsel * 8;
            const float g = acc[n][r];
            part = fmaf(label[(size_t)i * CDIM + j], f2v[r] + c2j - 2.0f * g, part);
        }
    }
    #pragma unroll
    for (int off = 16; off; off >>= 1) part += __shfl_xor(part, off);
    if (lane == 0) atomicAdd(out, part * SCALE);
}

// -------------------------------------------------------------------------
extern "C" void kernel_launch(void* const* d_in, const int* in_sizes, int n_in,
                              void* d_out, int out_size, void* d_ws, size_t ws_size,
                              hipStream_t stream) {
    (void)in_sizes; (void)n_in; (void)out_size; (void)ws_size;
    const float* feat    = (const float*)d_in[0];
    const float* label   = (const float*)d_in[1];
    const float* centers = (const float*)d_in[2];
    float* out = (float*)d_out;
    float* ws  = (float*)d_ws;          // [0,4096): f2   [4096,8192): c2

    // 8192 row-reductions, one wave each -> 1024 blocks of 256 threads
    center_loss_prep<<<1024, 256, 0, stream>>>(feat, centers, ws, out);

    dim3 grid(CDIM / 128, BDIM / 128);  // 32 x 32 macro-tiles
    center_loss_main<<<grid, 256, 0, stream>>>(feat, label, centers,
                                               ws, ws + BDIM, out);
}
